// TransformerImputer_34557306863839
// MI455X (gfx1250) — matrix operand order repaired and checked by reading, not compiled, not run on hardware
//
#include <hip/hip_runtime.h>
#include <hip/hip_bf16.h>

// ---------------------------------------------------------------- types
typedef __bf16 bf16;
typedef __attribute__((ext_vector_type(16))) __bf16        bf16x16;
typedef __attribute__((ext_vector_type(8)))  float         floatx8;
typedef __attribute__((ext_vector_type(4)))  float         floatx4;
typedef __attribute__((ext_vector_type(4)))  unsigned int  uintx4;
typedef __attribute__((ext_vector_type(8)))  unsigned int  uintx8;

__device__ __forceinline__ unsigned int pkbf(float a, float b) {
  union { bf16 h; unsigned short s; } ua, ub;
  ua.h = (bf16)a; ub.h = (bf16)b;
  return (unsigned int)ua.s | ((unsigned int)ub.s << 16);
}
// pack 8 floats -> 4 dwords of bf16 pairs (lowers to v_cvt_pk_bf16_f32)
__device__ __forceinline__ uintx4 pk8(floatx4 f0, floatx4 f1) {
  uintx4 u;
  u[0] = pkbf(f0[0], f0[1]); u[1] = pkbf(f0[2], f0[3]);
  u[2] = pkbf(f1[0], f1[1]); u[3] = pkbf(f1[2], f1[3]);
  return u;
}
// fragment = two 16B reads (LDS: ds_load_b128 x2, global: global_load_b128 x2)
__device__ __forceinline__ bf16x16 ldfrag(const bf16* p_lo, const bf16* p_hi) {
  uintx4 lo = *(const uintx4*)p_lo;
  uintx4 hi = *(const uintx4*)p_hi;
  uintx8 u = {lo[0], lo[1], lo[2], lo[3], hi[0], hi[1], hi[2], hi[3]};
  return __builtin_bit_cast(bf16x16, u);
}
__device__ __forceinline__ floatx8 wmma_bf16(bf16x16 a, bf16x16 b, floatx8 c) {
  return __builtin_amdgcn_wmma_f32_16x16x32_bf16(false, a, false, b, (short)0, c,
                                                 false, false);
}
// async global->LDS copy of 16B per lane (ASYNCcnt path); INST_OFFSET applies
// to both the LDS and global address, so parallel layouts share one base.
__device__ __forceinline__ void async_b128(unsigned lds_addr, const void* gptr,
                                           int /*compile-time*/ off0) {
  unsigned long long ga = (unsigned long long)(size_t)gptr;
  switch (off0) {
    case 0:
      asm volatile("global_load_async_to_lds_b128 %0, %1, off"
                   :: "v"(lds_addr), "v"(ga) : "memory");
      break;
    case 16:
      asm volatile("global_load_async_to_lds_b128 %0, %1, off offset:16"
                   :: "v"(lds_addr), "v"(ga) : "memory");
      break;
    case 32:
      asm volatile("global_load_async_to_lds_b128 %0, %1, off offset:32"
                   :: "v"(lds_addr), "v"(ga) : "memory");
      break;
    default:
      asm volatile("global_load_async_to_lds_b128 %0, %1, off offset:48"
                   :: "v"(lds_addr), "v"(ga) : "memory");
      break;
  }
}
__device__ __forceinline__ void wait_async0() {
  asm volatile("s_wait_asynccnt 0x0" ::: "memory");
}

// ---------------------------------------------------------------- constants
constexpr int D_MODEL = 1024;
constexpr int H_FF    = 1024;
constexpr int N_EXP   = 8;
constexpr int N_LAYER = 4;
constexpr int T_SEQ   = 512;
constexpr int N_BATCH = 4;
constexpr int N_TOK   = N_BATCH * T_SEQ;
constexpr int N_HEADS = 16;
constexpr int HD      = 64;
constexpr int KCHUNK  = 128;

enum { EPI_STORE = 0, EPI_ADD = 1, EPI_SILU = 2, EPI_MUL = 3, EPI_SCATTER = 4 };

// ============================================================== generic WMMA GEMM
// 128x64 block tile, 8 waves (4x2), each wave 32x32 -> 4 WMMA per 32-K step.
template <int LDA, int LDB, int LDC>
__global__ __launch_bounds__(256) void k_gemm(
    const float* __restrict__ A,
    const float* __restrict__ B,
    float* __restrict__ C,
    int M, int K,
    const int*   __restrict__ a_rows,
    const int*   __restrict__ m_count,
    const float* __restrict__ aux,
    const int*   __restrict__ c_rows,
    const float* __restrict__ row_w,
    int epi) {
  __shared__ __align__(16) bf16 As[128][40];  // [m][k] 128x32, stride 80B
  __shared__ __align__(16) bf16 Bs[64][40];   // [n][k] staged transposed

  const int Mact = m_count ? *m_count : M;
  const int row0 = blockIdx.y * 128;
  if (row0 >= Mact) return;                 // uniform early-exit (dynamic M)
  const int col0 = blockIdx.x * 64;

  const int tid  = threadIdx.x;
  const int wave = tid >> 5, lane = tid & 31;
  const int wm = wave >> 1, wn = wave & 1;  // 4x2 wave grid
  const int hi = lane >> 4, r = lane & 15;
  const int c0 = hi * 8;

  // ---- staging assignments (hoisted out of the K loop)
  const int rowl = tid >> 1, kb = (tid & 1) * 16;   // A: 16 floats along K
  const int growA = row0 + rowl;
  const bool aok  = growA < Mact;
  const int aidx  = aok ? growA : (Mact - 1);
  const int arow  = a_rows ? a_rows[aidx] : aidx;
  const float zm  = aok ? 1.f : 0.f;
  const float* Ap = A + (size_t)arow * LDA + kb;

  const int bn = tid & 63, bk = (tid >> 6) * 8;     // B: col bn, 8 k-values
  const float* Bp = B + col0 + bn;

  floatx8 acc00 = {}, acc01 = {}, acc10 = {}, acc11 = {};

  for (int k0 = 0; k0 < K; k0 += 32) {
    __syncthreads();
    { // stage A: 4x global_load_b128 -> pack -> 2x ds_store_b128
      floatx4 f0 = *(const floatx4*)(Ap + k0);
      floatx4 f1 = *(const floatx4*)(Ap + k0 + 4);
      floatx4 f2 = *(const floatx4*)(Ap + k0 + 8);
      floatx4 f3 = *(const floatx4*)(Ap + k0 + 12);
      f0 *= zm; f1 *= zm; f2 *= zm; f3 *= zm;
      *(uintx4*)&As[rowl][kb]     = pk8(f0, f1);
      *(uintx4*)&As[rowl][kb + 8] = pk8(f2, f3);
    }
    { // stage B transposed: 8x b32 (coalesced, immediate offsets) -> b128 store
      const float* bp = Bp + (size_t)(k0 + bk) * LDB;
      uintx4 u;
#pragma unroll
      for (int i = 0; i < 4; ++i)
        u[i] = pkbf(bp[(2 * i) * LDB], bp[(2 * i + 1) * LDB]);
      *(uintx4*)&Bs[bn][bk] = u;
    }
    if (k0 + 32 < K) {
      __builtin_prefetch(Ap + k0 + 32, 0, 1);
      __builtin_prefetch(Bp + (size_t)(k0 + bk + 32) * LDB, 0, 1);
    }
    __syncthreads();

    const bf16x16 af0 = ldfrag(&As[wm * 32 + r][c0],
                               &As[wm * 32 + r][c0 + 16]);
    const bf16x16 af1 = ldfrag(&As[wm * 32 + 16 + r][c0],
                               &As[wm * 32 + 16 + r][c0 + 16]);
    const bf16x16 b0  = ldfrag(&Bs[wn * 32 + r][c0],
                               &Bs[wn * 32 + r][c0 + 16]);
    const bf16x16 b1  = ldfrag(&Bs[wn * 32 + 16 + r][c0],
                               &Bs[wn * 32 + 16 + r][c0 + 16]);
    acc00 = wmma_bf16(af0, b0, acc00);
    acc01 = wmma_bf16(af0, b1, acc01);
    acc10 = wmma_bf16(af1, b0, acc10);
    acc11 = wmma_bf16(af1, b1, acc11);
  }

  // epilogue: C row = v + 8*hi within each 16-row tile, col = r within 16-col
#pragma unroll
  for (int rt = 0; rt < 2; ++rt) {
    const floatx8 aA = rt ? acc10 : acc00;
    const floatx8 aB = rt ? acc11 : acc01;
#pragma unroll
    for (int v = 0; v < 8; ++v) {
      const int grow = row0 + wm * 32 + rt * 16 + hi * 8 + v;
      if (grow >= Mact) continue;
      const int gc0 = col0 + wn * 32 + r;
      const int gc1 = gc0 + 16;
      float a0 = aA[v], a1 = aB[v];
      if (epi == EPI_STORE) {
        C[(size_t)grow * LDC + gc0] = a0;
        C[(size_t)grow * LDC + gc1] = a1;
      } else if (epi == EPI_ADD) {
        C[(size_t)grow * LDC + gc0] += a0;
        C[(size_t)grow * LDC + gc1] += a1;
      } else if (epi == EPI_SILU) {
        C[(size_t)grow * LDC + gc0] = a0 / (1.f + __expf(-a0));
        C[(size_t)grow * LDC + gc1] = a1 / (1.f + __expf(-a1));
      } else if (epi == EPI_MUL) {
        const size_t o = (size_t)grow * LDC;
        C[o + gc0] = a0 * aux[o + gc0];
        C[o + gc1] = a1 * aux[o + gc1];
      } else { // EPI_SCATTER: rows unique within launch -> plain RMW
        const int cr   = c_rows[grow];
        const float w  = row_w[grow];
        const size_t o = (size_t)cr * LDC;
        C[o + gc0] += w * a0;
        C[o + gc1] += w * a1;
      }
    }
  }
}

// ============================================================== flash attention
// One workgroup per (b,h). Q/K come from the RoPE'd bf16 mirror (Q via direct
// global b128 fragment loads, K via global_load_async_to_lds_b128); V is
// transposed+converted from the f32 qkv. Diagonal-only mask, online softmax.
__global__ __launch_bounds__(256) void k_attention(const float* __restrict__ qkv,
                                                   const bf16* __restrict__ qkvb,
                                                   float* __restrict__ y) {
  const int bh = blockIdx.x;
  const int b = bh / N_HEADS, h = bh % N_HEADS;
  const int tid  = threadIdx.x;
  const int wave = tid >> 5, lane = tid & 31;
  const int hi = lane >> 4, r = lane & 15;
  const int c0 = hi * 8;

  __shared__ __align__(16) bf16 Ks[KCHUNK][72];   // [key][dim]  stride 144B
  __shared__ __align__(16) bf16 Vt[HD][136];      // [dim][key]  stride 272B
  __shared__ __align__(16) bf16 Ps[8][16][40];    // per-wave P tile

  const float scale = 0.125f;                     // 1/sqrt(64)

  for (int qg = 0; qg < 4; ++qg) {
    const int qt = qg * 8 + wave;                 // q tile 0..31
    const int q0 = qt * 16;

    // Q fragments: two global b128 loads each from the bf16 mirror
    const bf16* Qb = qkvb + (size_t)(b * T_SEQ + q0 + r) * (3 * D_MODEL) +
                     h * HD;
    const bf16x16 aq0 = ldfrag(Qb + c0, Qb + c0 + 16);
    const bf16x16 aq1 = ldfrag(Qb + 32 + c0, Qb + 32 + c0 + 16);

    floatx8 O0 = {}, O1 = {}, O2 = {}, O3 = {};
    float mrow[8], lrow[8];
#pragma unroll
    for (int v = 0; v < 8; ++v) { mrow[v] = -1e30f; lrow[v] = 0.f; }

    for (int ch = 0; ch < T_SEQ / KCHUNK; ++ch) {
      const int kb_g = ch * KCHUNK;
      __syncthreads();
      { // stage K: async global(bf16)->LDS, 4x 16B per lane; the LDS and
        // global offsets within one half-row match, so one base + offsets.
        const int key = tid >> 1, hb = (tid & 1) * 32;  // half-row of 32 dims
        const bf16* gk = qkvb + (size_t)(b * T_SEQ + kb_g + key) *
                         (3 * D_MODEL) + D_MODEL + h * HD + hb;
        const unsigned la = (unsigned)(size_t)&Ks[key][hb];
        async_b128(la, gk, 0);
        async_b128(la, gk, 16);
        async_b128(la, gk, 32);
        async_b128(la, gk, 48);
      }
      { // stage V transposed from f32: lane owns one dim (coalesced)
        const int dim = tid & 63, kg = (tid >> 6) * 32;
        const float* Vb = qkv + (size_t)(b * T_SEQ + kb_g + kg) * (3 * D_MODEL) +
                          2 * D_MODEL + h * HD + dim;
#pragma unroll
        for (int g = 0; g < 4; ++g) {
          uintx4 u;
#pragma unroll
          for (int i = 0; i < 4; ++i)
            u[i] = pkbf(Vb[(size_t)(8 * g + 2 * i) * (3 * D_MODEL)],
                        Vb[(size_t)(8 * g + 2 * i + 1) * (3 * D_MODEL)]);
          *(uintx4*)&Vt[dim][kg + 8 * g] = u;
        }
      }
      wait_async0();        // ASYNCcnt -> 0 before the workgroup barrier
      __syncthreads();

      for (int kk = 0; kk < KCHUNK; kk += 32) {
        const bf16x16 bkA0 = ldfrag(&Ks[kk + r][c0],
                                    &Ks[kk + r][c0 + 16]);
        const bf16x16 bkA1 = ldfrag(&Ks[kk + r][32 + c0],
                                    &Ks[kk + r][32 + c0 + 16]);
        const bf16x16 bkB0 = ldfrag(&Ks[kk + 16 + r][c0],
                                    &Ks[kk + 16 + r][c0 + 16]);
        const bf16x16 bkB1 = ldfrag(&Ks[kk + 16 + r][32 + c0],
                                    &Ks[kk + 16 + r][32 + c0 + 16]);
        floatx8 s0 = {}, s1 = {};
        s0 = wmma_bf16(aq0, bkA0, s0);
        s0 = wmma_bf16(aq1, bkA1, s0);
        s1 = wmma_bf16(aq0, bkB0, s1);
        s1 = wmma_bf16(aq1, bkB1, s1);

        // online softmax update; stash P into per-wave LDS tile
#pragma unroll
        for (int v = 0; v < 8; ++v) {
          const int qi  = q0 + v + 8 * hi;
          const int k0i = kb_g + kk + r;
          float e0 = s0[v] * scale + (qi == k0i      ? -1e9f : 0.f);
          float e1 = s1[v] * scale + (qi == k0i + 16 ? -1e9f : 0.f);
          float tmax = fmaxf(e0, e1);
          tmax = fmaxf(tmax, __shfl_xor(tmax, 1, 32));
          tmax = fmaxf(tmax, __shfl_xor(tmax, 2, 32));
          tmax = fmaxf(tmax, __shfl_xor(tmax, 4, 32));
          tmax = fmaxf(tmax, __shfl_xor(tmax, 8, 32));
          const float mnew  = fmaxf(mrow[v], tmax);
          const float alpha = __expf(mrow[v] - mnew);
          const float p0 = __expf(e0 - mnew), p1 = __expf(e1 - mnew);
          float ps = p0 + p1;
          ps += __shfl_xor(ps, 1, 32);
          ps += __shfl_xor(ps, 2, 32);
          ps += __shfl_xor(ps, 4, 32);
          ps += __shfl_xor(ps, 8, 32);
          lrow[v] = lrow[v] * alpha + ps;
          mrow[v] = mnew;
          Ps[wave][v + 8 * hi][r]      = (bf16)p0;
          Ps[wave][v + 8 * hi][16 + r] = (bf16)p1;
          O0[v] *= alpha; O1[v] *= alpha; O2[v] *= alpha; O3[v] *= alpha;
        }

        // P as A-operand (wave-private round trip; DScnt ordering per wave)
        const bf16x16 ap = ldfrag(&Ps[wave][r][c0], &Ps[wave][r][c0 + 16]);
        const bf16x16 bv0 = ldfrag(&Vt[r][kk + c0],
                                   &Vt[r][kk + c0 + 16]);
        const bf16x16 bv1 = ldfrag(&Vt[16 + r][kk + c0],
                                   &Vt[16 + r][kk + c0 + 16]);
        const bf16x16 bv2 = ldfrag(&Vt[32 + r][kk + c0],
                                   &Vt[32 + r][kk + c0 + 16]);
        const bf16x16 bv3 = ldfrag(&Vt[48 + r][kk + c0],
                                   &Vt[48 + r][kk + c0 + 16]);
        O0 = wmma_bf16(ap, bv0, O0);
        O1 = wmma_bf16(ap, bv1, O1);
        O2 = wmma_bf16(ap, bv2, O2);
        O3 = wmma_bf16(ap, bv3, O3);
      }
    }

    // normalize and write y[token][h*64 + dim]
#pragma unroll
    for (int v = 0; v < 8; ++v) {
      const int qi = q0 + v + 8 * hi;
      const float inv = 1.f / lrow[v];
      float* yr = y + (size_t)(b * T_SEQ + qi) * D_MODEL + h * HD;
      yr[r]      = O0[v] * inv;
      yr[16 + r] = O1[v] * inv;
      yr[32 + r] = O2[v] * inv;
      yr[48 + r] = O3[v] * inv;
    }
  }
}

// ============================================================== small kernels
__global__ __launch_bounds__(256) void k_embed(const float* __restrict__ x,
                                               const float* __restrict__ w_in,
                                               float* __restrict__ hx) {
  const int idx = blockIdx.x * 256 + threadIdx.x;
  if (idx >= N_TOK * D_MODEL) return;
  const int j = idx & (D_MODEL - 1), t = idx >> 10;
  float s = 0.f;
#pragma unroll
  for (int f = 0; f < 8; ++f) s += x[t * 8 + f] * w_in[f * D_MODEL + j];
  hx[idx] = s;
}

__global__ __launch_bounds__(256) void k_rmsnorm(const float* __restrict__ xin,
                                                 const float* __restrict__ w,
                                                 float* __restrict__ o) {
  __shared__ float red[9];
  const int row = blockIdx.x;
  const float* xr = xin + (size_t)row * D_MODEL;
  float ss = 0.f;
  for (int i = threadIdx.x; i < D_MODEL; i += 256) { float v = xr[i]; ss += v * v; }
  ss += __shfl_xor(ss, 1, 32);
  ss += __shfl_xor(ss, 2, 32);
  ss += __shfl_xor(ss, 4, 32);
  ss += __shfl_xor(ss, 8, 32);
  ss += __shfl_xor(ss, 16, 32);
  if ((threadIdx.x & 31) == 0) red[threadIdx.x >> 5] = ss;
  __syncthreads();
  if (threadIdx.x == 0) {
    float t2 = 0.f;
#pragma unroll
    for (int i = 0; i < 8; ++i) t2 += red[i];
    red[8] = rsqrtf(t2 / (float)D_MODEL + 1e-6f);
  }
  __syncthreads();
  const float rs = red[8];
  for (int i = threadIdx.x; i < D_MODEL; i += 256)
    o[(size_t)row * D_MODEL + i] = xr[i] * rs * w[i];
}

// RoPE: read f32 q/k from qkv, rotate, write packed bf16 pairs to the mirror.
__global__ __launch_bounds__(256) void k_rope(const float* __restrict__ qkv,
                                              bf16* __restrict__ qkvb) {
  const int idx = blockIdx.x * 256 + threadIdx.x;
  if (idx >= N_TOK * N_HEADS * 32) return;
  const int pair = idx & 31;
  const int tmp  = idx >> 5;
  const int h    = tmp & (N_HEADS - 1);
  const int t    = tmp >> 4;
  const int tl   = t & (T_SEQ - 1);
  const float freq = __powf(10000.f, -((2.f * pair) / (float)HD));
  const float ang  = (float)tl * freq;
  float c, s;
  __sincosf(ang, &c, &s);
  const size_t base = (size_t)t * (3 * D_MODEL) + h * HD + 2 * pair;
  unsigned int* qb32 = (unsigned int*)qkvb;
  const float q1 = qkv[base], q2 = qkv[base + 1];
  qb32[base >> 1] = pkbf(q1 * c - q2 * s, q1 * s + q2 * c);
  const float k1 = qkv[base + D_MODEL], k2 = qkv[base + D_MODEL + 1];
  qb32[(base + D_MODEL) >> 1] = pkbf(k1 * c - k2 * s, k1 * s + k2 * c);
}

__global__ void k_zero_int(int* __restrict__ p, int n) {
  const int i = blockIdx.x * blockDim.x + threadIdx.x;
  if (i < n) p[i] = 0;
}

// router: one wave per token; softmax over 8 experts, top-2 scatter into lists
__global__ __launch_bounds__(256) void k_router(const float* __restrict__ hn,
                                                const float* __restrict__ wr,
                                                int* __restrict__ idx_list,
                                                float* __restrict__ w_list,
                                                int* __restrict__ cnt) {
  const int wave = threadIdx.x >> 5, lane = threadIdx.x & 31;
  const int t = blockIdx.x * 8 + wave;
  if (t >= N_TOK) return;
  float s[N_EXP];
#pragma unroll
  for (int e = 0; e < N_EXP; ++e) s[e] = 0.f;
  const float* hr = hn + (size_t)t * D_MODEL;
  for (int dd = lane; dd < D_MODEL; dd += 32) {
    const float hv = hr[dd];
    const float* wrow = wr + (size_t)dd * N_EXP;
#pragma unroll
    for (int e = 0; e < N_EXP; ++e) s[e] += hv * wrow[e];
  }
#pragma unroll
  for (int e = 0; e < N_EXP; ++e)
    for (int m = 16; m >= 1; m >>= 1) s[e] += __shfl_xor(s[e], m, 32);
  float mx = s[0];
#pragma unroll
  for (int e = 1; e < N_EXP; ++e) mx = fmaxf(mx, s[e]);
  float p[N_EXP], sum = 0.f;
#pragma unroll
  for (int e = 0; e < N_EXP; ++e) { p[e] = __expf(s[e] - mx); sum += p[e]; }
#pragma unroll
  for (int e = 0; e < N_EXP; ++e) p[e] /= sum;
  int i0 = 0;
#pragma unroll
  for (int e = 1; e < N_EXP; ++e) if (p[e] > p[i0]) i0 = e;
  int i1 = (i0 == 0) ? 1 : 0;
#pragma unroll
  for (int e = 0; e < N_EXP; ++e) if (e != i0 && p[e] > p[i1]) i1 = e;
  if (lane == 0) {
    const int o0 = atomicAdd(&cnt[i0], 1);
    idx_list[i0 * N_TOK + o0] = t; w_list[i0 * N_TOK + o0] = p[i0];
    const int o1 = atomicAdd(&cnt[i1], 1);
    idx_list[i1 * N_TOK + o1] = t; w_list[i1 * N_TOK + o1] = p[i1];
  }
}

__global__ __launch_bounds__(256) void k_outproj(const float* __restrict__ hn,
                                                 const float* __restrict__ w_out,
                                                 float* __restrict__ out) {
  const int idx = blockIdx.x * 256 + threadIdx.x;
  if (idx >= N_TOK * 8) return;
  const int f = idx & 7, t = idx >> 3;
  float s = 0.f;
  const float* hr = hn + (size_t)t * D_MODEL;
  for (int dd = 0; dd < D_MODEL; ++dd) s += hr[dd] * w_out[(size_t)dd * 8 + f];
  out[idx] = s;
}

// ============================================================== orchestration
extern "C" void kernel_launch(void* const* d_in, const int* in_sizes, int n_in,
                              void* d_out, int out_size, void* d_ws, size_t ws_size,
                              hipStream_t stream) {
  (void)in_sizes; (void)n_in; (void)out_size; (void)ws_size;
  const float* x        = (const float*)d_in[0];
  const float* w_in     = (const float*)d_in[1];
  const float* w_out    = (const float*)d_in[2];
  const float* w_final  = (const float*)d_in[3];
  const float* rms1     = (const float*)d_in[4];
  const float* rms2     = (const float*)d_in[5];
  const float* wqkv     = (const float*)d_in[6];
  const float* wo       = (const float*)d_in[7];
  const float* w_router = (const float*)d_in[8];
  const float* w_fc     = (const float*)d_in[9];
  const float* w_gate   = (const float*)d_in[10];
  const float* w_proj   = (const float*)d_in[11];
  const float* sh_fc    = (const float*)d_in[12];
  const float* sh_gate  = (const float*)d_in[13];
  const float* sh_proj  = (const float*)d_in[14];

  constexpr int d = D_MODEL, hf = H_FF;
  float* ws   = (float*)d_ws;
  float* hx   = ws;
  float* hn   = hx + (size_t)N_TOK * d;
  float* y    = hn + (size_t)N_TOK * d;
  float* qkv  = y + (size_t)N_TOK * d;
  float* gb   = qkv + (size_t)N_TOK * 3 * d;
  float* fb   = gb + (size_t)N_TOK * hf;
  float* eg   = fb + (size_t)N_TOK * hf;
  float* ef   = eg + (size_t)N_TOK * hf;
  float* wlst = ef + (size_t)N_TOK * hf;
  int* ilst   = (int*)(wlst + (size_t)N_EXP * N_TOK);
  int* cnt    = ilst + (size_t)N_EXP * N_TOK;
  bf16* qkvb  = (bf16*)(cnt + 64);              // bf16 mirror of rope'd q/k

  const dim3 blk(256);
  k_embed<<<(N_TOK * d + 255) / 256, blk, 0, stream>>>(x, w_in, hx);

  for (int l = 0; l < N_LAYER; ++l) {
    // --- attention block
    k_rmsnorm<<<N_TOK, blk, 0, stream>>>(hx, rms1 + (size_t)l * d, hn);
    {
      dim3 g(3 * d / 64, N_TOK / 128);
      k_gemm<1024, 3072, 3072><<<g, blk, 0, stream>>>(
          hn, wqkv + (size_t)l * d * 3 * d, qkv, N_TOK, d,
          nullptr, nullptr, nullptr, nullptr, nullptr, EPI_STORE);
    }
    k_rope<<<(N_TOK * N_HEADS * 32 + 255) / 256, blk, 0, stream>>>(qkv, qkvb);
    k_attention<<<N_BATCH * N_HEADS, blk, 0, stream>>>(qkv, qkvb, y);
    {
      dim3 g(d / 64, N_TOK / 128);
      k_gemm<1024, 1024, 1024><<<g, blk, 0, stream>>>(
          y, wo + (size_t)l * d * d, hx, N_TOK, d,
          nullptr, nullptr, nullptr, nullptr, nullptr, EPI_ADD);
    }

    // --- MoE block
    k_rmsnorm<<<N_TOK, blk, 0, stream>>>(hx, rms2 + (size_t)l * d, hn);
    k_zero_int<<<1, 32, 0, stream>>>(cnt, N_EXP);
    k_router<<<N_TOK / 8, blk, 0, stream>>>(hn, w_router + (size_t)l * d * N_EXP,
                                            ilst, wlst, cnt);
    for (int e = 0; e < N_EXP; ++e) {
      const float* wg = w_gate + ((size_t)l * N_EXP + e) * d * hf;
      const float* wf = w_fc   + ((size_t)l * N_EXP + e) * d * hf;
      const float* wp = w_proj + ((size_t)l * N_EXP + e) * hf * d;
      dim3 gf(hf / 64, N_TOK / 128);
      k_gemm<1024, 1024, 1024><<<gf, blk, 0, stream>>>(
          hn, wg, eg, N_TOK, d,
          ilst + e * N_TOK, cnt + e, nullptr, nullptr, nullptr, EPI_SILU);
      k_gemm<1024, 1024, 1024><<<gf, blk, 0, stream>>>(
          hn, wf, ef, N_TOK, d,
          ilst + e * N_TOK, cnt + e, eg, nullptr, nullptr, EPI_MUL);
      dim3 gp(d / 64, N_TOK / 128);
      k_gemm<1024, 1024, 1024><<<gp, blk, 0, stream>>>(
          ef, wp, hx, N_TOK, hf,
          nullptr, cnt + e, nullptr, ilst + e * N_TOK, wlst + e * N_TOK,
          EPI_SCATTER);
    }
    // --- shared MLP
    {
      dim3 gf(hf / 64, N_TOK / 128);
      k_gemm<1024, 1024, 1024><<<gf, blk, 0, stream>>>(
          hn, sh_gate + (size_t)l * d * hf, gb, N_TOK, d,
          nullptr, nullptr, nullptr, nullptr, nullptr, EPI_SILU);
      k_gemm<1024, 1024, 1024><<<gf, blk, 0, stream>>>(
          hn, sh_fc + (size_t)l * d * hf, fb, N_TOK, d,
          nullptr, nullptr, gb, nullptr, nullptr, EPI_MUL);
      dim3 gp(d / 64, N_TOK / 128);
      k_gemm<1024, 1024, 1024><<<gp, blk, 0, stream>>>(
          fb, sh_proj + (size_t)l * hf * d, hx, N_TOK, hf,
          nullptr, nullptr, nullptr, nullptr, nullptr, EPI_ADD);
    }
  }

  k_rmsnorm<<<N_TOK, blk, 0, stream>>>(hx, w_final, hn);
  k_outproj<<<(N_TOK * 8 + 255) / 256, blk, 0, stream>>>(hn, w_out, (float*)d_out);
}
